// LIFRouterLearnableBeta_24249385353829
// MI455X (gfx1250) — matrix operand order repaired and checked by reading, not compiled, not run on hardware
//
#include <hip/hip_runtime.h>
#include <hip/hip_bf16.h>

typedef __attribute__((ext_vector_type(16))) _Float16 v16h;
typedef __attribute__((ext_vector_type(8)))  _Float16 v8h;
typedef __attribute__((ext_vector_type(2)))  __fp16   h2v;   // matches __builtin_amdgcn_cvt_pkrtz return type
typedef __attribute__((ext_vector_type(8)))  float    v8f;
typedef __attribute__((ext_vector_type(4)))  float    v4f;   // clang vector (works with nontemporal builtins)

static constexpr int kB = 16;
static constexpr int kT = 4096;
static constexpr int kD = 1024;
static constexpr int kE = 64;
static constexpr int kRowsPerBlock = 128;     // 8 waves * 16 rows
static constexpr int kChunk = 64;             // scan chunk length ((a,c,d) state handles it exactly)
static constexpr int kNChunk = kT / kChunk;   // 64
static constexpr int kPitch = 66;             // LDS pitch in floats (bank-conflict padding)
static constexpr int kKSteps = kD / 32;       // 32

// ---------------- Kernel 0: W (f32) -> f16 once ----------------
__global__ void w_to_f16(const float* __restrict__ W, _Float16* __restrict__ Wh, int n) {
    int i = blockIdx.x * blockDim.x + threadIdx.x;
    if (i < n) Wh[i] = (_Float16)W[i];
}

// ---------------- Kernel 1: fused GEMM (WMMA f16) + per-chunk LIF state ----------------
// Each block: 128 consecutive timesteps of one batch row. Each wave: 16 t-rows x all 64 experts.
__global__ __launch_bounds__(256)
void lif_gemm_chunkscan(const float* __restrict__ seq,
                        const _Float16* __restrict__ Wh,
                        const float* __restrict__ bias,
                        const float* __restrict__ beta_raw,
                        float* __restrict__ gA, float* __restrict__ gC, float* __restrict__ gD)
{
    __shared__ float Ilds[kRowsPerBlock * kPitch];   // 33792 B

    const int tid   = threadIdx.x;
    const int wave  = tid >> 5;
    const int lane  = tid & 31;
    const int lhalf = lane & 15;   // row-in-tile / col-in-tile
    const int g     = lane >> 4;   // half-wave selector

    const int m0  = blockIdx.x * kRowsPerBlock;          // global row = b*T + t
    const int row = m0 + wave * 16 + lhalf;              // this lane's A row
    const v4f* __restrict__ arow = (const v4f*)(seq + (size_t)row * kD);
    const _Float16* __restrict__ wrow = Wh + (size_t)lhalf * kD + 8 * g;  // + nt*16*kD + 32*kk

    // Accumulators: 4 N-tiles of 16; bias folded in (C element r -> col = nt*16+lhalf for all r)
    v8f acc[4];
#pragma unroll
    for (int nt = 0; nt < 4; ++nt) {
        float bv = bias[nt * 16 + lhalf];
#pragma unroll
        for (int r = 0; r < 8; ++r) acc[nt][r] = bv;
    }

    union AFrag { v16h v; h2v h[8]; };

    // ---- software pipeline: double-buffered A (f32) and B (f16) fragments ----
    // A lane layout (16-bit A 16x32): elements 0..7 <- K=8g+0..7 ; elements 8..15 <- K=16+8g+0..7
    // B lane layout (32x16):          col N = lhalf, elements 0..7 <- K=8g+0..7 ; elements 8..15 <- K=16+8g+0..7 (same as A)
    v4f aF[4];
    v8h bF[8];
    {
        aF[0] = __builtin_nontemporal_load(&arow[2 * g + 0]);
        aF[1] = __builtin_nontemporal_load(&arow[2 * g + 1]);
        aF[2] = __builtin_nontemporal_load(&arow[2 * g + 4]);
        aF[3] = __builtin_nontemporal_load(&arow[2 * g + 5]);
#pragma unroll
        for (int nt = 0; nt < 4; ++nt) {
            const _Float16* wp = wrow + (size_t)nt * 16 * kD;
            bF[2 * nt + 0] = *(const v8h*)wp;
            bF[2 * nt + 1] = *(const v8h*)(wp + 16);
        }
    }

#pragma unroll 2
    for (int kk = 0; kk < kKSteps; ++kk) {
        // Prefetch the streamed seq row 8 K-steps (1KB) ahead (global_prefetch_b8; speculative, OOB-safe)
        __builtin_prefetch((const void*)(arow + 8 * (kk + 8)), 0, 1);

        // ---- issue next iteration's loads first (redundant reload of last tile on final iter) ----
        const int kn = (kk + 1 < kKSteps) ? (kk + 1) : kk;
        v4f aN[4];
        v8h bN[8];
        aN[0] = __builtin_nontemporal_load(&arow[8 * kn + 2 * g + 0]);
        aN[1] = __builtin_nontemporal_load(&arow[8 * kn + 2 * g + 1]);
        aN[2] = __builtin_nontemporal_load(&arow[8 * kn + 2 * g + 4]);
        aN[3] = __builtin_nontemporal_load(&arow[8 * kn + 2 * g + 5]);
#pragma unroll
        for (int nt = 0; nt < 4; ++nt) {
            const _Float16* wp = wrow + (size_t)nt * 16 * kD + 32 * kn;
            bN[2 * nt + 0] = *(const v8h*)wp;
            bN[2 * nt + 1] = *(const v8h*)(wp + 16);
        }

        // ---- convert A to f16 fragment ----
        AFrag af;
        af.h[0] = __builtin_amdgcn_cvt_pkrtz(aF[0].x, aF[0].y);
        af.h[1] = __builtin_amdgcn_cvt_pkrtz(aF[0].z, aF[0].w);
        af.h[2] = __builtin_amdgcn_cvt_pkrtz(aF[1].x, aF[1].y);
        af.h[3] = __builtin_amdgcn_cvt_pkrtz(aF[1].z, aF[1].w);
        af.h[4] = __builtin_amdgcn_cvt_pkrtz(aF[2].x, aF[2].y);
        af.h[5] = __builtin_amdgcn_cvt_pkrtz(aF[2].z, aF[2].w);
        af.h[6] = __builtin_amdgcn_cvt_pkrtz(aF[3].x, aF[3].y);
        af.h[7] = __builtin_amdgcn_cvt_pkrtz(aF[3].z, aF[3].w);

        // ---- 4 WMMAs (one per 16-wide N tile) ----
#pragma unroll
        for (int nt = 0; nt < 4; ++nt) {
            v16h bf = __builtin_shufflevector(bF[2 * nt], bF[2 * nt + 1],
                                              0,1,2,3,4,5,6,7,8,9,10,11,12,13,14,15);
            acc[nt] = __builtin_amdgcn_wmma_f32_16x16x32_f16(
                false, af.v, false, bf, (short)0, acc[nt], false, false);
        }

        // ---- rotate buffers (with unroll 2 these become pure renames) ----
#pragma unroll
        for (int j = 0; j < 4; ++j) aF[j] = aN[j];
#pragma unroll
        for (int j = 0; j < 8; ++j) bF[j] = bN[j];
    }

    // ---- Stage I tile to LDS. C/D layout: element r of lane l -> (M = r + 8g, N = lhalf)
#pragma unroll
    for (int nt = 0; nt < 4; ++nt) {
        const int e = nt * 16 + lhalf;
#pragma unroll
        for (int r = 0; r < 8; ++r) {
            const int lrow = wave * 16 + r + 8 * g;
            Ilds[lrow * kPitch + e] = acc[nt][r];
        }
    }
    __syncthreads();

    // ---- Per-chunk LIF state: f(u)=min(a*u+c, d). Compose 64 steps: a'=b*a, c'=b*c+I, d'=min(b*d+I,1).
    if (tid < 128) {
        const int e  = tid & 63;
        const int ci = tid >> 6;     // which of the two 64-step chunks in this block
        const float beta = 1.0f / (1.0f + expf(-beta_raw[e]));
        float a = 1.0f, c = 0.0f, d = 1e30f;
        const float* col = Ilds + (ci * kChunk) * kPitch + e;
#pragma unroll 1
        for (int i = 0; i < kChunk; ++i) {
            const float I = col[i * kPitch];
            c = beta * c + I;
            d = fminf(beta * d + I, 1.0f);
            a = beta * a;
        }
        const int b          = m0 >> 12;                        // / kT
        const int chunk_in_b = ((m0 & (kT - 1)) >> 6) + ci;     // t0/64 + ci
        const int idx        = chunk_in_b * (kB * kE) + b * kE + e;
        gA[idx] = a; gC[idx] = c; gD[idx] = d;
    }
}

// ---------------- Kernel 2: compose chunk states in t-order, then softmax over E ----------------
__global__ __launch_bounds__(1024)
void lif_finalize(const float* __restrict__ gA, const float* __restrict__ gC,
                  const float* __restrict__ gD, float* __restrict__ out)
{
    __shared__ float Ush[kB * kE];
    const int tid = threadIdx.x;              // tid = b*64 + e
    float u = 0.0f;
#pragma unroll 1
    for (int ci = 0; ci < kNChunk; ++ci) {
        const int idx = ci * (kB * kE) + tid;
        u = fminf(gA[idx] * u + gC[idx], gD[idx]);
    }
    Ush[tid] = u;
    __syncthreads();
    const int b = tid >> 6;
    float mx = -3.4e38f;
#pragma unroll 1
    for (int j = 0; j < kE; ++j) mx = fmaxf(mx, Ush[b * kE + j]);
    float s = 0.0f;
#pragma unroll 1
    for (int j = 0; j < kE; ++j) s += expf(Ush[b * kE + j] - mx);
    out[tid] = expf(u - mx) / s;
}

extern "C" void kernel_launch(void* const* d_in, const int* in_sizes, int n_in,
                              void* d_out, int out_size, void* d_ws, size_t ws_size,
                              hipStream_t stream) {
    const float* seq  = (const float*)d_in[0];   // [16,4096,1024]
    const float* W    = (const float*)d_in[1];   // [64,1024]
    const float* bias = (const float*)d_in[2];   // [64]
    const float* braw = (const float*)d_in[3];   // [64]

    char* ws = (char*)d_ws;
    _Float16* Wh = (_Float16*)ws;                          // 131072 B
    float* gA = (float*)(ws + (size_t)kE * kD * sizeof(_Float16));
    float* gC = gA + kNChunk * kB * kE;                    // 65536 floats each
    float* gD = gC + kNChunk * kB * kE;                    // total ws use ~0.9 MB

    w_to_f16<<<(kE * kD + 255) / 256, 256, 0, stream>>>(W, Wh, kE * kD);
    lif_gemm_chunkscan<<<(kB * kT) / kRowsPerBlock, 256, 0, stream>>>(
        seq, Wh, bias, braw, gA, gC, gD);
    lif_finalize<<<1, kB * kE, 0, stream>>>(gA, gC, gD, (float*)d_out);
}
